// AttnDecoderRNN_31568009626350
// MI455X (gfx1250) — matrix operand order repaired and checked by reading, not compiled, not run on hardware
//
#include <hip/hip_runtime.h>
#include <hip/hip_bf16.h>
#include <cmath>

// CDNA5 (gfx1250) wave32 WMMA types
typedef __attribute__((ext_vector_type(16))) __bf16 v16bf;
typedef __attribute__((ext_vector_type(8)))  float  v8f;

#define TPB 256

__device__ __forceinline__ unsigned short bf16_bits(float f) {
  union { __bf16 b; unsigned short s; } u;
  u.b = (__bf16)f;                       // native v_cvt path on gfx1250
  return u.s;
}

// ---------------- f32 -> bf16 staging for GEMV vectors ----------------
__global__ void k_cvt_bf16(const float* __restrict__ in, unsigned short* __restrict__ ob, int n) {
  int i = blockIdx.x * blockDim.x + threadIdx.x;
  if (i < n) ob[i] = bf16_bits(in[i]);
}

// ---------------- generic GEMV via v_wmma_f32_16x16x32_bf16 ----------------
// out[r] = dot(W[r,:], vec) + bias[r]   (optionally relu'd)
// One wave per 16-row tile. A = 16x32 fp32->bf16 tile of W, B = vec broadcast into
// all 16 columns, C accumulates f32. Fragment layouts used here (contiguous K, same order in A and B):
//   A lane<16: row=lane, K=kc+0..15; lane>=16: K=kc+16..31
//   B lanes 0-15: K=kc+0..15 ; lanes 16-31: K=kc+16..31 (columns identical)
//   C lane 0 -> rows 0..7, lane 16 -> rows 8..15
__global__ void k_gemv_wmma(const float* __restrict__ W, const unsigned short* __restrict__ vb,
                            const float* __restrict__ bias, float* __restrict__ out,
                            int rows, int K, int relu) {
  int wave = (int)((blockIdx.x * (unsigned)blockDim.x + threadIdx.x) >> 5);
  int lane = threadIdx.x & 31;
  int row0 = wave << 4;
  if (row0 >= rows) return;              // wave-uniform: EXEC stays all-ones for WMMA
  int m  = lane & 15;
  int hi = lane >> 4;                    // 0: K low half, 1: K high half
  int row = row0 + m; if (row >= rows) row = rows - 1;   // clamp tail reads
  const float* wr = W + (size_t)row * K;
  int ko = hi * 16;
  v8f c = {};
#pragma unroll 4
  for (int kc = 0; kc < K; kc += 32) {
    __builtin_prefetch(wr + kc + 32, 0, 0);              // speculative, drop-on-fault
    const v8f f0 = *(const v8f*)(wr + kc + ko);          // 32B-aligned b128 pair
    const v8f f1 = *(const v8f*)(wr + kc + ko + 8);
    v16bf a;
#pragma unroll
    for (int i = 0; i < 8; ++i) { a[i] = (__bf16)f0[i]; a[8 + i] = (__bf16)f1[i]; }
    v16bf b = *(const v16bf*)(vb + kc + (hi << 4));      // 32B-aligned
    c = __builtin_amdgcn_wmma_f32_16x16x32_bf16(false, a, false, b, (short)0, c, false, false);
  }
  if (m == 0) {
    int mb = hi * 8;
#pragma unroll
    for (int p = 0; p < 8; ++p) {
      int r = row0 + mb + p;
      if (r < rows) {
        float v = c[p] + (bias ? bias[r] : 0.f);
        out[r] = relu ? fmaxf(v, 0.f) : v;
      }
    }
  }
}

// ---------------- softmax over L (single block, blockDim = 512) ----------------
__global__ void k_softmax(float* __restrict__ sc, float* __restrict__ out_attn, int L) {
  __shared__ float sm[512];
  int t = threadIdx.x;
  float v = (t < L) ? sc[t] : -INFINITY;
  sm[t] = v; __syncthreads();
  for (int o = 256; o > 0; o >>= 1) {
    if (t < o) sm[t] = fmaxf(sm[t], sm[t + o]);
    __syncthreads();
  }
  float m = sm[0]; __syncthreads();
  float e = (t < L) ? __expf(v - m) : 0.f;
  sm[t] = e; __syncthreads();
  for (int o = 256; o > 0; o >>= 1) {
    if (t < o) sm[t] += sm[t + o];
    __syncthreads();
  }
  float inv = 1.f / sm[0];
  if (t < L) { float w = e * inv; sc[t] = w; out_attn[t] = w; }
}

// ---------------- combined = [embed(tok), attn_weights @ enc] ----------------
__global__ void k_combine(const float* __restrict__ emb, const int* __restrict__ tok,
                          const float* __restrict__ w, const float* __restrict__ enc,
                          float* __restrict__ comb, int H, int L) {
  int i = blockIdx.x * blockDim.x + threadIdx.x;
  if (i >= 2 * H) return;
  if (i < H) {
    comb[i] = emb[(size_t)tok[0] * H + i];
  } else {
    int j = i - H;
    float s = 0.f;
    for (int l = 0; l < L; ++l) s = fmaf(w[l], enc[(size_t)l * H + j], s);
    comb[i] = s;
  }
}

// ---------------- GRU elementwise update ----------------
__global__ void k_gru(const float* __restrict__ gi, const float* __restrict__ gh,
                      const float* __restrict__ h0, float* __restrict__ hn,
                      unsigned short* __restrict__ hb, float* __restrict__ hout, int H) {
  int j = blockIdx.x * blockDim.x + threadIdx.x;
  if (j >= H) return;
  float r = 1.f / (1.f + __expf(-(gi[j] + gh[j])));
  float z = 1.f / (1.f + __expf(-(gi[H + j] + gh[H + j])));
  float n = tanhf(gi[2 * H + j] + r * gh[2 * H + j]);
  float h = (1.f - z) * n + z * h0[j];
  hn[j] = h;
  hout[j] = h;
  hb[j] = bf16_bits(h);
}

// ---------------- log-softmax: partial max -> (max, log-sum-exp) -> logp ----------------
__global__ void k_maxpart(const float* __restrict__ logits, float* __restrict__ bmax, int V) {
  __shared__ float sm[TPB];
  float m = -INFINITY;
  for (int i = blockIdx.x * TPB + threadIdx.x; i < V; i += gridDim.x * TPB)
    m = fmaxf(m, logits[i]);
  sm[threadIdx.x] = m; __syncthreads();
  for (int o = TPB / 2; o > 0; o >>= 1) {
    if (threadIdx.x < o) sm[threadIdx.x] = fmaxf(sm[threadIdx.x], sm[threadIdx.x + o]);
    __syncthreads();
  }
  if (threadIdx.x == 0) bmax[blockIdx.x] = sm[0];
}

__global__ void k_stats(const float* __restrict__ logits, const float* __restrict__ bmax,
                        float* __restrict__ stats, int V, int NB) {
  __shared__ float sm[TPB];
  float m = -INFINITY;
  for (int i = threadIdx.x; i < NB; i += TPB) m = fmaxf(m, bmax[i]);
  sm[threadIdx.x] = m; __syncthreads();
  for (int o = TPB / 2; o > 0; o >>= 1) {
    if (threadIdx.x < o) sm[threadIdx.x] = fmaxf(sm[threadIdx.x], sm[threadIdx.x + o]);
    __syncthreads();
  }
  float gmax = sm[0]; __syncthreads();
  float s = 0.f;
  for (int i = threadIdx.x; i < V; i += TPB) s += __expf(logits[i] - gmax);
  sm[threadIdx.x] = s; __syncthreads();
  for (int o = TPB / 2; o > 0; o >>= 1) {
    if (threadIdx.x < o) sm[threadIdx.x] += sm[threadIdx.x + o];
    __syncthreads();
  }
  if (threadIdx.x == 0) { stats[0] = gmax; stats[1] = logf(sm[0]); }
}

__global__ void k_logp(const float* __restrict__ logits, const float* __restrict__ stats,
                       float* __restrict__ out, int V) {
  int i = blockIdx.x * blockDim.x + threadIdx.x;
  if (i < V) out[i] = logits[i] - stats[0] - stats[1];
}

static inline void launch_gemv(const float* W, const unsigned short* vb, const float* bias,
                               float* out, int rows, int K, int relu, hipStream_t stream) {
  int waves  = (rows + 15) / 16;     // one wave per 16 output rows
  int blocks = (waves + 7) / 8;      // 8 waves (256 threads) per block
  k_gemv_wmma<<<blocks, TPB, 0, stream>>>(W, vb, bias, out, rows, K, relu);
}

extern "C" void kernel_launch(void* const* d_in, const int* in_sizes, int n_in,
                              void* d_out, int out_size, void* d_ws, size_t ws_size,
                              hipStream_t stream) {
  const int*   tok  = (const int*)d_in[0];
  const float* h0   = (const float*)d_in[1];
  const float* enc  = (const float*)d_in[2];
  const float* embw = (const float*)d_in[3];
  const float* acw  = (const float*)d_in[4];
  const float* acb  = (const float*)d_in[5];
  const float* wih  = (const float*)d_in[6];
  const float* whh  = (const float*)d_in[7];
  const float* bih  = (const float*)d_in[8];
  const float* bhh  = (const float*)d_in[9];
  const float* outw = (const float*)d_in[10];
  const float* outb = (const float*)d_in[11];

  const int H = in_sizes[1];           // 1024
  const int L = in_sizes[2] / H;       // 512
  const int V = in_sizes[11];          // 50257

  float* out      = (float*)d_out;
  float* out_logp = out;               // V
  float* out_h    = out + V;           // H
  float* out_attn = out + V + H;       // L

  // workspace layout (floats); all bf16 blocks land on 32B boundaries
  float* ws    = (float*)d_ws;
  float* sc    = ws;                                   // L
  float* comb  = sc + L;                               // 2H
  float* x     = comb + 2 * H;                         // H
  float* gi    = x + H;                                // 3H
  float* gh    = gi + 3 * H;                           // 3H
  float* hn    = gh + 3 * H;                           // H
  unsigned short* h0b   = (unsigned short*)(hn + H);           // H u16
  unsigned short* combb = h0b + H;                              // 2H u16
  unsigned short* xb    = combb + 2 * H;                        // H u16
  unsigned short* hnb   = xb + H;                               // H u16
  float* logits = (float*)(hnb + H);                   // V
  float* bmax   = logits + V;                          // 128
  float* stats  = bmax + 128;                          // 2

  // 1) stage h0 in bf16; attention scores = enc @ h0  (GEMV, rows=L, K=H)
  k_cvt_bf16<<<(H + TPB - 1) / TPB, TPB, 0, stream>>>(h0, h0b, H);
  launch_gemv(enc, h0b, nullptr, sc, L, H, 0, stream);

  // 2) softmax -> attn weights (also to output)
  k_softmax<<<1, 512, 0, stream>>>(sc, out_attn, L);

  // 3) combined = [embedding row, attn @ enc]
  k_combine<<<(2 * H + TPB - 1) / TPB, TPB, 0, stream>>>(embw, tok, sc, enc, comb, H, L);
  k_cvt_bf16<<<(2 * H + TPB - 1) / TPB, TPB, 0, stream>>>(comb, combb, 2 * H);

  // 4) x = relu(attn_combine_w @ combined + b)   (rows=H, K=2H)
  launch_gemv(acw, combb, acb, x, H, 2 * H, 1, stream);
  k_cvt_bf16<<<(H + TPB - 1) / TPB, TPB, 0, stream>>>(x, xb, H);

  // 5) GRU gates: gi = w_ih @ x + b_ih ; gh = w_hh @ h0 + b_hh   (rows=3H, K=H)
  launch_gemv(wih, xb, bih, gi, 3 * H, H, 0, stream);
  launch_gemv(whh, h0b, bhh, gh, 3 * H, H, 0, stream);

  // 6) GRU elementwise -> h_new (f32 out + bf16 staging)
  k_gru<<<(H + TPB - 1) / TPB, TPB, 0, stream>>>(gi, gh, h0, hn, hnb, out_h, H);

  // 7) logits = out_w @ h_new + out_b   (rows=V, K=H) — dominant 206 MB stream
  launch_gemv(outw, hnb, outb, logits, V, H, 0, stream);

  // 8) log-softmax
  k_maxpart<<<128, TPB, 0, stream>>>(logits, bmax, V);
  k_stats<<<1, TPB, 0, stream>>>(logits, bmax, stats, V, 128);
  k_logp<<<(V + TPB - 1) / TPB, TPB, 0, stream>>>(logits, stats, out_logp, V);
}
